// ConvNet_54374285967584
// MI455X (gfx1250) — matrix operand order repaired and checked by reading, not compiled, not run on hardware
//
#include <hip/hip_runtime.h>
#include <hip/hip_bf16.h>

// ---------------------------------------------------------------------------
// Types for CDNA5 WMMA (wave32): D(f32 16x16) = A(bf16 16x32) * B(bf16 32x16) + C
// ---------------------------------------------------------------------------
typedef __attribute__((ext_vector_type(16))) __bf16 v16bf;
typedef __attribute__((ext_vector_type(8)))  float  v8f;

#define N_WORDS 4096
#define L_CHARS 32
#define DIM     512
#define XT_STRIDE 528   // padded row stride (bf16 elems): 1056B -> 32B aligned rows, reduced bank conflicts

__device__ __forceinline__ unsigned short f2bf(float f) {
    unsigned u = __float_as_uint(f);
    u = (u + 0x7FFFu + ((u >> 16) & 1u)) >> 16;   // RNE
    return (unsigned short)u;
}
__device__ __forceinline__ unsigned fkey(float f) {  // monotone float -> uint
    int i = __float_as_int(f);
    unsigned u = (unsigned)i;
    return (i >= 0) ? (u | 0x80000000u) : ~u;
}
__device__ __forceinline__ float keyf(unsigned k) {  // inverse
    unsigned u = (k & 0x80000000u) ? (k & 0x7FFFFFFFu) : ~k;
    return __uint_as_float(u);
}

// ---------------------------------------------------------------------------
// Prep: bf16 char-embedding table with padding row 0 zeroed
// ---------------------------------------------------------------------------
__global__ void prep_ce_kernel(const float* __restrict__ chr_emb,
                               unsigned short* __restrict__ ce_bf) {
    int idx = blockIdx.x * 256 + threadIdx.x;          // 256*512 elems
    if (idx >= 256 * DIM) return;
    int row = idx >> 9;
    float v = (row == 0) ? 0.0f : chr_emb[idx];
    ce_bf[idx] = f2bf(v);
}

// ---------------------------------------------------------------------------
// Pack conv_chr_w [512][512][3] into per-lane A-fragment layout.
// Tile id = (mt*3 + t)*16 + ks ; tile = 32 lanes x 16 bf16.
// Contiguous A layout (matches B): lane<16 -> K in {0..15}; lane>=16 -> {16..31}
// ---------------------------------------------------------------------------
__global__ void pack_conv_kernel(const float* __restrict__ w,
                                 unsigned short* __restrict__ packed) {
    int idx = blockIdx.x * 256 + threadIdx.x;          // 1536*512 elems
    if (idx >= 1536 * 512) return;
    int tile = idx >> 9, within = idx & 511;
    int lane = within >> 4, q = within & 15;
    int mt = tile / 48, rest = tile % 48;
    int t = rest / 16, ks = rest % 16;
    int m = lane & 15, kh = lane >> 4;
    int K = kh * 16 + q;
    int o = mt * 16 + m;
    int i = ks * 32 + K;
    packed[idx] = f2bf(w[(o * DIM + i) * 3 + t]);
}

// Pack conv_sent_w[:, :, 1] [1024][1024] the same way. Tile id = ot*32 + ks.
__global__ void pack_sent_kernel(const float* __restrict__ w,
                                 unsigned short* __restrict__ packed) {
    int idx = blockIdx.x * 256 + threadIdx.x;          // 2048*512 elems
    if (idx >= 2048 * 512) return;
    int tile = idx >> 9, within = idx & 511;
    int lane = within >> 4, q = within & 15;
    int ot = tile / 32, ks = tile % 32;
    int m = lane & 15, kh = lane >> 4;
    int K = kh * 16 + q;
    int o = ot * 16 + m;
    int i = ks * 32 + K;
    packed[idx] = f2bf(w[(o * 1024 + i) * 3 + 1]);
}

// ---------------------------------------------------------------------------
// Word-embedding gather -> u_bf16[n][0..511]  (padding_idx = 0)
// ---------------------------------------------------------------------------
__global__ void gather_word_kernel(const int* __restrict__ words,
                                   const float* __restrict__ word_emb,
                                   unsigned short* __restrict__ ub) {
    int n = blockIdx.x;
    int wi = words[n];
    for (int d = threadIdx.x; d < DIM; d += 256) {
        float v = (wi == 0) ? 0.0f : word_emb[(size_t)wi * DIM + d];
        ub[(size_t)n * 1024 + d] = f2bf(v);
    }
}

// ---------------------------------------------------------------------------
// Char conv as 3 tap-shifted GEMMs + max-pool over l, all via bf16 WMMA.
// One workgroup (8 waves) per word. Each wave: 4 M-tiles x 2 N-tiles.
// ---------------------------------------------------------------------------
__global__ void __launch_bounds__(256)
conv_wmma_kernel(const int* __restrict__ words_in_char,
                 const unsigned short* __restrict__ ce_bf,
                 const unsigned short* __restrict__ packedW,
                 const float* __restrict__ conv_b,
                 unsigned short* __restrict__ ub) {
    __shared__ unsigned short xt[33 * XT_STRIDE];  // XT[l][i], row 32 = zeros (OOB taps)
    __shared__ int wcs[L_CHARS];

    const int n = blockIdx.x;
    const int tid = threadIdx.x;
    const int wave = tid >> 5, lane = tid & 31;
    const int m16 = lane & 15, kh = lane >> 4;

    if (tid < L_CHARS) wcs[tid] = words_in_char[n * L_CHARS + tid];
    __syncthreads();

    // Build XT: XT[l][16g + j] = ce[wc[g]][32j + l]  (the "reshape" semantics)
    for (int idx = tid; idx < L_CHARS * DIM; idx += 256) {
        int g = idx >> 9, rem = idx & 511;
        int j = rem >> 5, l = rem & 31;
        xt[l * XT_STRIDE + g * 16 + j] = ce_bf[wcs[g] * DIM + rem];
    }
    for (int idx = tid; idx < XT_STRIDE; idx += 256)
        xt[32 * XT_STRIDE + idx] = 0;   // zero row for tap boundary
    __syncthreads();

    v8f acc[4][2];
    #pragma unroll
    for (int mt = 0; mt < 4; ++mt)
        #pragma unroll
        for (int nt = 0; nt < 2; ++nt)
            #pragma unroll
            for (int e = 0; e < 8; ++e) acc[mt][nt][e] = 0.0f;

    for (int t = 0; t < 3; ++t) {
        for (int ks = 0; ks < 16; ++ks) {
            v16bf a[4];
            #pragma unroll
            for (int mt = 0; mt < 4; ++mt) {
                int tile = ((wave * 4 + mt) * 3 + t) * 16 + ks;
                a[mt] = *(const v16bf*)(packedW + (size_t)tile * 512 + lane * 16);
            }
            #pragma unroll
            for (int nt = 0; nt < 2; ++nt) {
                int c = nt * 16 + m16 + t - 1;          // shifted source column
                int row = (c >= 0 && c < L_CHARS) ? c : 32;  // OOB -> zero row
                const v16bf b = *(const v16bf*)(xt + row * XT_STRIDE + ks * 32 + kh * 16);
                #pragma unroll
                for (int mt = 0; mt < 4; ++mt)
                    acc[mt][nt] = __builtin_amdgcn_wmma_f32_16x16x32_bf16(
                        false, a[mt], false, b, (short)0, acc[mt][nt], false, false);
            }
        }
    }

    // Max over l (columns): C layout puts N=l across lanes within each 16-lane half.
    #pragma unroll
    for (int mt = 0; mt < 4; ++mt) {
        float mx[8];
        #pragma unroll
        for (int r = 0; r < 8; ++r) mx[r] = fmaxf(acc[mt][0][r], acc[mt][1][r]);
        #pragma unroll
        for (int r = 0; r < 8; ++r) {
            mx[r] = fmaxf(mx[r], __shfl_xor(mx[r], 1, 32));
            mx[r] = fmaxf(mx[r], __shfl_xor(mx[r], 2, 32));
            mx[r] = fmaxf(mx[r], __shfl_xor(mx[r], 4, 32));
            mx[r] = fmaxf(mx[r], __shfl_xor(mx[r], 8, 32));
        }
        if (m16 == 0) {   // lane 0 holds M=r, lane 16 holds M=r+8
            int obase = (wave * 4 + mt) * 16 + kh * 8;
            #pragma unroll
            for (int r = 0; r < 8; ++r) {
                float v = mx[r] + conv_b[obase + r];
                ub[(size_t)n * 1024 + 512 + obase + r] = f2bf(v);
            }
        }
    }
}

// ---------------------------------------------------------------------------
// r = u @ W1c^T + b, max over words, via bf16 WMMA + atomic key-max.
// grid: x = word-block (32 words), y = o-block (128 o). 1 M-tile per wave.
// ---------------------------------------------------------------------------
__global__ void init_rkey_kernel(unsigned* __restrict__ rkey) {
    int i = blockIdx.x * 256 + threadIdx.x;
    if (i < 1024) rkey[i] = 0u;   // minimal key = identity for max
}

__global__ void __launch_bounds__(256)
sent_wmma_kernel(const unsigned short* __restrict__ ub,
                 const unsigned short* __restrict__ packedS,
                 const float* __restrict__ sent_b,
                 unsigned* __restrict__ rkey) {
    const int tid = threadIdx.x;
    const int wave = tid >> 5, lane = tid & 31;
    const int m16 = lane & 15, kh = lane >> 4;
    const int ot = blockIdx.y * 8 + wave;                 // o tile in [0,64)
    const int w0 = blockIdx.x * 32 + m16;                 // word column (nt=0)

    v8f acc0, acc1;
    #pragma unroll
    for (int e = 0; e < 8; ++e) { acc0[e] = 0.0f; acc1[e] = 0.0f; }

    for (int ks = 0; ks < 32; ++ks) {
        v16bf a = *(const v16bf*)(packedS + ((size_t)ot * 32 + ks) * 512 + lane * 16);
        const v16bf b0 = *(const v16bf*)(ub + (size_t)w0 * 1024 + ks * 32 + kh * 16);
        const v16bf b1 = *(const v16bf*)(ub + (size_t)(w0 + 16) * 1024 + ks * 32 + kh * 16);
        acc0 = __builtin_amdgcn_wmma_f32_16x16x32_bf16(false, a, false, b0, (short)0, acc0, false, false);
        acc1 = __builtin_amdgcn_wmma_f32_16x16x32_bf16(false, a, false, b1, (short)0, acc1, false, false);
    }

    float mx[8];
    #pragma unroll
    for (int r = 0; r < 8; ++r) mx[r] = fmaxf(acc0[r], acc1[r]);
    #pragma unroll
    for (int r = 0; r < 8; ++r) {
        mx[r] = fmaxf(mx[r], __shfl_xor(mx[r], 1, 32));
        mx[r] = fmaxf(mx[r], __shfl_xor(mx[r], 2, 32));
        mx[r] = fmaxf(mx[r], __shfl_xor(mx[r], 4, 32));
        mx[r] = fmaxf(mx[r], __shfl_xor(mx[r], 8, 32));
    }
    if (m16 == 0) {
        int obase = ot * 16 + kh * 8;
        #pragma unroll
        for (int r = 0; r < 8; ++r) {
            float v = mx[r] + sent_b[obase + r];
            atomicMax(&rkey[obase + r], fkey(v));
        }
    }
}

// ---------------------------------------------------------------------------
// Head (exact f32): h = tanh(r @ lin1_w^T + b1); out = h @ lin2_w^T + b2
// Single 256-thread workgroup.
// ---------------------------------------------------------------------------
__global__ void __launch_bounds__(256)
head_kernel(const unsigned* __restrict__ rkey,
            const float* __restrict__ lin1_w, const float* __restrict__ lin1_b,
            const float* __restrict__ lin2_w, const float* __restrict__ lin2_b,
            float* __restrict__ out) {
    __shared__ float rf[1024];
    __shared__ float h[2048];
    __shared__ float red[256];
    const int tid = threadIdx.x;

    for (int i = tid; i < 1024; i += 256) rf[i] = keyf(rkey[i]);
    __syncthreads();

    for (int j = 0; j < 8; ++j) {
        int o = j * 256 + tid;
        float s = lin1_b[o];
        const float* wrow = lin1_w + (size_t)o * 1024;
        for (int k = 0; k < 1024; ++k) s = fmaf(wrow[k], rf[k], s);
        h[o] = tanhf(s);
    }
    __syncthreads();

    for (int o2 = 0; o2 < 2; ++o2) {
        float s = 0.0f;
        const float* wrow = lin2_w + (size_t)o2 * 2048;
        for (int k = tid; k < 2048; k += 256) s = fmaf(wrow[k], h[k], s);
        red[tid] = s;
        __syncthreads();
        for (int stride = 128; stride > 0; stride >>= 1) {
            if (tid < stride) red[tid] += red[tid + stride];
            __syncthreads();
        }
        if (tid == 0) out[o2] = red[0] + lin2_b[o2];
        __syncthreads();
    }
}

// ---------------------------------------------------------------------------
// Launch
// ---------------------------------------------------------------------------
extern "C" void kernel_launch(void* const* d_in, const int* in_sizes, int n_in,
                              void* d_out, int out_size, void* d_ws, size_t ws_size,
                              hipStream_t stream) {
    (void)in_sizes; (void)n_in; (void)out_size; (void)ws_size;
    const int*   words         = (const int*)  d_in[0];
    const int*   words_in_char = (const int*)  d_in[1];
    const float* word_emb      = (const float*)d_in[2];
    const float* chr_emb       = (const float*)d_in[3];
    const float* conv_chr_w    = (const float*)d_in[4];
    const float* conv_chr_b    = (const float*)d_in[5];
    const float* conv_sent_w   = (const float*)d_in[6];
    const float* conv_sent_b   = (const float*)d_in[7];
    const float* lin1_w        = (const float*)d_in[8];
    const float* lin1_b        = (const float*)d_in[9];
    const float* lin2_w        = (const float*)d_in[10];
    const float* lin2_b        = (const float*)d_in[11];
    float* out = (float*)d_out;

    char* ws = (char*)d_ws;
    unsigned short* ce_bf    = (unsigned short*)(ws + 0);          // 256*512*2      = 262144
    unsigned short* packedW  = (unsigned short*)(ws + 262144);     // 1536*512*2     = 1572864
    unsigned short* packedS  = (unsigned short*)(ws + 1835008);    // 2048*512*2     = 2097152
    unsigned short* ub       = (unsigned short*)(ws + 3932160);    // 4096*1024*2    = 8388608
    unsigned*       rkey     = (unsigned*)      (ws + 12320768);   // 1024*4

    prep_ce_kernel   <<<512,  256, 0, stream>>>(chr_emb, ce_bf);
    pack_conv_kernel <<<3072, 256, 0, stream>>>(conv_chr_w, packedW);
    pack_sent_kernel <<<4096, 256, 0, stream>>>(conv_sent_w, packedS);
    gather_word_kernel<<<N_WORDS, 256, 0, stream>>>(words, word_emb, ub);

    conv_wmma_kernel <<<N_WORDS, 256, 0, stream>>>(words_in_char, ce_bf, packedW,
                                                   conv_chr_b, ub);

    init_rkey_kernel <<<4, 256, 0, stream>>>(rkey);
    dim3 sg(N_WORDS / 32, 8);   // 128 word-blocks x 8 o-blocks
    sent_wmma_kernel <<<sg, 256, 0, stream>>>(ub, packedS, conv_sent_b, rkey);

    head_kernel      <<<1, 256, 0, stream>>>(rkey, lin1_w, lin1_b, lin2_w, lin2_b, out);
}
